// RNN_80272938762395
// MI455X (gfx1250) — compile-verified, matrix operand order checked by reading, not run on hardware
//
#include <hip/hip_runtime.h>
#include <hip/hip_bf16.h>

// ---------------- problem constants ----------------
#define BATCH 256
#define TT    1024
#define INF   302          // input feature dim
#define KIN   320          // padded to multiple of 32
#define HID   512
#define GATES 1536         // 3*HID
#define NTILES 96          // GATES/16
#define KT_IH 10           // KIN/32
#define KT_HH 16           // HID/32
#define MTILES 16384       // BATCH*TT/16

typedef __attribute__((ext_vector_type(16))) __bf16 v16bf;
typedef __attribute__((ext_vector_type(8)))  float  v8f;
typedef unsigned int u32x4 __attribute__((ext_vector_type(4)));
typedef int          i32x4 __attribute__((ext_vector_type(4)));
typedef int          i32x8 __attribute__((ext_vector_type(8)));

#if defined(__HIP_DEVICE_COMPILE__) && __has_builtin(__builtin_amdgcn_tensor_load_to_lds)
#define HAVE_TDM 1
#else
#define HAVE_TDM 0
#endif

__device__ __forceinline__ unsigned short f2bf(float f) {
    unsigned u = __float_as_uint(f);
    unsigned r = (u + 0x7FFFu + ((u >> 16) & 1u)) >> 16;   // RNE
    return (unsigned short)r;
}

// k-index pattern of element j inside a 16-bit WMMA fragment for lane L
// (ISA 7.12.2: j=0..7 -> K=base+j, j=8..15 -> K=16+base+(j-8), base=(L>>4)*8)
__device__ __forceinline__ int klocal(int lane, int j) {
    int base = (lane >> 4) * 8;
    return (j < 8) ? (base + j) : (16 + base + (j - 8));
}

#if HAVE_TDM
// 1-D contiguous TDM load: nelem f32 elements from gptr -> LDS at lds_ptr.
// D# per CDNA5 ISA: group0 {count=1, lds_addr, global_addr, type=2},
// group1 {data_size=4B, tensor_dim0=tile_dim0=nelem, tensor_dim1=1, dim0_stride=nelem}.
__device__ __forceinline__ void tdm_load_1d(void* lds_ptr, const float* gptr, int nelem) {
    unsigned lds_off = (unsigned)(uintptr_t)lds_ptr;            // low 32 bits = LDS offset
    unsigned long long ga = (unsigned long long)(uintptr_t)gptr;
    u32x4 g0;
    g0[0] = 1u;                                                 // count=1 (valid), user mode
    g0[1] = lds_off;                                            // bits[63:32]  lds_addr
    g0[2] = (unsigned)(ga & 0xFFFFFFFFu);                       // bits[95:64]  global_addr lo
    g0[3] = (unsigned)((ga >> 32) & 0x01FFFFFFu) | 0x80000000u; // addr hi | type=2 (bits127:126)
    i32x8 g1;
    g1[0] = 0x00020000;                        // wg_mask=0, data_size=2 (4 bytes)
    g1[1] = (int)((nelem & 0xFFFF) << 16);     // tensor_dim0[15:0] at bits[63:48]
    g1[2] = (int)(((unsigned)nelem >> 16) | (1u << 16)); // tensor_dim0[31:16] | tensor_dim1=1
    g1[3] = (int)((nelem & 0xFFFF) << 16);     // tile_dim0 at bits[127:112]
    g1[4] = 0;                                 // tile_dim1=0, tile_dim2=0 (unused)
    g1[5] = nelem;                             // tensor_dim0_stride lo32
    g1[6] = 0; g1[7] = 0;
    i32x4 z4 = {0, 0, 0, 0};
#if __clang_major__ >= 23
    i32x8 z8 = {0, 0, 0, 0, 0, 0, 0, 0};
    __builtin_amdgcn_tensor_load_to_lds(g0, g1, z4, z4, z8, 0);
#else
    __builtin_amdgcn_tensor_load_to_lds(g0, g1, z4, z4, 0);
#endif
}
#endif

// ---------------- pack kernels (f32 -> fragment-native bf16) ----------------

__global__ void pack_x_kernel(const float* __restrict__ x, unsigned short* __restrict__ xpk) {
    long tid = (long)blockIdx.x * blockDim.x + threadIdx.x;
    if (tid >= (long)MTILES * KT_IH * 512) return;
    int j    = (int)(tid & 15);
    int lane = (int)((tid >> 4) & 31);
    int kt   = (int)((tid >> 9) % KT_IH);
    int mt   = (int)(tid / (512 * KT_IH));
    int m = lane & 15;
    int k = kt * 32 + klocal(lane, j);
    int R = mt * 16 + m;           // flattened (b,t): b outer
    int b = R >> 10, t = R & 1023;
    float v = (k < INF) ? x[((size_t)b * TT + t) * INF + k] : 0.f;
    xpk[(((size_t)mt * KT_IH + kt) * 32 + lane) * 16 + j] = f2bf(v);
}

__global__ void pack_w_kernel(const float* __restrict__ w, unsigned short* __restrict__ wpk,
                              int ktiles, int Ksrc) {
    long tid = (long)blockIdx.x * blockDim.x + threadIdx.x;
    if (tid >= (long)ktiles * NTILES * 512) return;
    int j    = (int)(tid & 15);
    int lane = (int)((tid >> 4) & 31);
    int nt   = (int)((tid >> 9) % NTILES);
    int kt   = (int)(tid / (512 * NTILES));
    int n = nt * 16 + (lane & 15);
    int k = kt * 32 + klocal(lane, j);
    float v = (k < Ksrc) ? w[(size_t)n * Ksrc + k] : 0.f;
    wpk[(((size_t)kt * NTILES + nt) * 32 + lane) * 16 + j] = f2bf(v);
}

// ---------------- kernel 1: ig = x @ w_ih^T + b  (bf16 WMMA, f32 out) ----------------
// one wave computes a 16(M) x 128(N) strip: 8 accumulators, B loads batched 4-at-a-time
// so the scheduler can clause 8x global_load_b128 ahead of each WMMA group.
__global__ void __launch_bounds__(256)
ig_gemm_kernel(const unsigned short* __restrict__ xpk,
               const unsigned short* __restrict__ wihpk,
               const float* __restrict__ bias,
               float* __restrict__ ig) {
    int lane = threadIdx.x & 31;
    int gw   = blockIdx.x * 8 + (threadIdx.x >> 5);
    int mt   = gw / 12;
    int n0   = (gw % 12) * 8;

    v8f acc[8] = {};
    #pragma unroll
    for (int kt = 0; kt < KT_IH; ++kt) {
        union { v16bf v; uint4 q[2]; } A;
        const uint4* ap = (const uint4*)(xpk + (((size_t)mt * KT_IH + kt) * 32 + lane) * 16);
        A.q[0] = ap[0]; A.q[1] = ap[1];
        #pragma unroll
        for (int g = 0; g < 2; ++g) {
            union { v16bf v; uint4 q[2]; } B0, B1, B2, B3;
            const uint4* bp0 = (const uint4*)(wihpk + (((size_t)kt * NTILES + n0 + 4 * g + 0) * 32 + lane) * 16);
            const uint4* bp1 = (const uint4*)(wihpk + (((size_t)kt * NTILES + n0 + 4 * g + 1) * 32 + lane) * 16);
            const uint4* bp2 = (const uint4*)(wihpk + (((size_t)kt * NTILES + n0 + 4 * g + 2) * 32 + lane) * 16);
            const uint4* bp3 = (const uint4*)(wihpk + (((size_t)kt * NTILES + n0 + 4 * g + 3) * 32 + lane) * 16);
            B0.q[0] = bp0[0]; B0.q[1] = bp0[1];
            B1.q[0] = bp1[0]; B1.q[1] = bp1[1];
            B2.q[0] = bp2[0]; B2.q[1] = bp2[1];
            B3.q[0] = bp3[0]; B3.q[1] = bp3[1];
            acc[4 * g + 0] = __builtin_amdgcn_wmma_f32_16x16x32_bf16(false, A.v, false, B0.v, (short)0, acc[4 * g + 0], false, false);
            acc[4 * g + 1] = __builtin_amdgcn_wmma_f32_16x16x32_bf16(false, A.v, false, B1.v, (short)0, acc[4 * g + 1], false, false);
            acc[4 * g + 2] = __builtin_amdgcn_wmma_f32_16x16x32_bf16(false, A.v, false, B2.v, (short)0, acc[4 * g + 2], false, false);
            acc[4 * g + 3] = __builtin_amdgcn_wmma_f32_16x16x32_bf16(false, A.v, false, B3.v, (short)0, acc[4 * g + 3], false, false);
        }
    }
    #pragma unroll
    for (int i = 0; i < 8; ++i) {
        int n = (n0 + i) * 16 + (lane & 15);
        float bv = bias[n];
        #pragma unroll
        for (int r = 0; r < 8; ++r) {
            int m = r + ((lane >> 4) << 3);
            int R = mt * 16 + m;
            int b = R >> 10, t = R & 1023;
            ig[((size_t)t * BATCH + b) * GATES + n] = acc[i][r] + bv;
        }
    }
}

// ---------------- kernel 2: persistent GRU recurrence + fused tail ----------------
// 16 blocks x 512 threads (16 waves). Block owns batch rows [b0, b0+16).
// h in LDS (bf16 WMMA copy + f32 copy); ig[t] rows TDM-prefetched into LDS double buffer.
__global__ void __launch_bounds__(512)
gru_recurrent_kernel(const unsigned short* __restrict__ whhpk,
                     const float* __restrict__ ig,
                     const float* __restrict__ b_n,
                     const float* __restrict__ ln_w,
                     const float* __restrict__ ln_b,
                     const float* __restrict__ w1,
                     const float* __restrict__ b1,
                     const float* __restrict__ w2,
                     const float* __restrict__ out_bias,
                     float* __restrict__ out) {
    __shared__ __align__(16) unsigned short hbf[16][520];  // bf16 h (padded vs bank conflicts)
    __shared__ __align__(16) float           hf [16][520]; // f32 h
#if HAVE_TDM
    __shared__ __align__(16) float igbuf[2][16][GATES];    // TDM double buffer (2 x 96 KB)
    float (*h1s)[256] = (float(*)[256])&igbuf[0][0][0];    // reused after recurrence
#else
    __shared__ float h1s[16][256];
#endif

    const int w    = threadIdx.x >> 5;
    const int lane = threadIdx.x & 31;
    const int b0   = blockIdx.x * 16;

    for (int i = threadIdx.x; i < 16 * 520; i += 512) {
        ((unsigned short*)hbf)[i] = 0;
        ((float*)hf)[i] = 0.f;
    }
#if HAVE_TDM
    if (w == 0) {  // prologue: bring ig[t=0] rows (contiguous 96 KB) into buffer 0
        if (lane == 0) tdm_load_1d(&igbuf[0][0][0], ig + (size_t)b0 * GATES, 16 * GATES);
        __builtin_amdgcn_s_wait_tensorcnt(0);
    }
#endif
    __syncthreads();

    const float bn0 = b_n[w * 32 + (lane & 15)];
    const float bn1 = b_n[w * 32 + 16 + (lane & 15)];
    const int nt[6] = { 2 * w, 2 * w + 1, 32 + 2 * w, 33 + 2 * w, 64 + 2 * w, 65 + 2 * w };

    for (int t = 0; t < TT; ++t) {
#if HAVE_TDM
        if (w == 0 && lane == 0 && t + 1 < TT)   // overlap next step's ig DMA with WMMA phase
            tdm_load_1d(&igbuf[(t + 1) & 1][0][0],
                        ig + ((size_t)(t + 1) * BATCH + b0) * GATES, 16 * GATES);
#endif
        v8f acc[6] = {};
        #pragma unroll 4
        for (int kt = 0; kt < KT_HH; ++kt) {
            // A fragment = two contiguous 16B chunks of the bf16 h row
            union { v16bf v; uint4 q[2]; } A;
            const int m = lane & 15, base = (lane >> 4) * 8;
            A.q[0] = *(const uint4*)&hbf[m][kt * 32 + base];
            A.q[1] = *(const uint4*)&hbf[m][kt * 32 + 16 + base];
            if (kt + 1 < KT_HH)  // keep w_hh L2 stream ahead (global_prefetch_b8)
                __builtin_prefetch(whhpk + (((size_t)(kt + 1) * NTILES + nt[0]) * 32 + lane) * 16, 0, 1);
            #pragma unroll
            for (int q = 0; q < 6; ++q) {
                union { v16bf v; uint4 q4[2]; } Bm;
                const uint4* bp = (const uint4*)(whhpk + (((size_t)kt * NTILES + nt[q]) * 32 + lane) * 16);
                Bm.q4[0] = bp[0]; Bm.q4[1] = bp[1];
                acc[q] = __builtin_amdgcn_wmma_f32_16x16x32_bf16(false, A.v, false, Bm.v,
                                                                 (short)0, acc[q], false, false);
            }
        }
        __syncthreads();   // all hbf reads done before updates

        #pragma unroll
        for (int st = 0; st < 2; ++st) {
            const int j = w * 32 + st * 16 + (lane & 15);
            const float bnv = st ? bn1 : bn0;
            #pragma unroll
            for (int r = 0; r < 8; ++r) {
                int m = r + ((lane >> 4) << 3);
#if HAVE_TDM
                const float* igp = &igbuf[t & 1][m][0];
#else
                const float* igp = ig + ((size_t)t * BATCH + b0 + m) * GATES;
#endif
                float ir = igp[j], iz = igp[HID + j], inn = igp[2 * HID + j];
                float rr = 1.f / (1.f + __expf(-(ir + acc[0 + st][r])));
                float zz = 1.f / (1.f + __expf(-(iz + acc[2 + st][r])));
                float nn = tanhf(inn + rr * (acc[4 + st][r] + bnv));
                float ho = hf[m][j];
                float hn2 = nn + zz * (ho - nn);
                hf[m][j]  = hn2;
                hbf[m][j] = f2bf(hn2);
            }
        }
#if HAVE_TDM
        if (w == 0) __builtin_amdgcn_s_wait_tensorcnt(0);  // publish buf (t+1)&1 before barrier
#endif
        __syncthreads();
    }

    // ---- fused tail: dropout -> LayerNorm -> 512x256 -> 256x2 -> sigmoid ----
    {
        float sum = 0.f, sumsq = 0.f;
        for (int c = lane; c < HID; c += 32) {
            float v = hf[w][c];
            unsigned hsh = ((unsigned)(b0 + w) * HID + (unsigned)c) * 2654435761u + 12345u;
            v = ((hsh >> 16) & 1u) ? v * 2.0f : 0.f;  // dropout p=0.5, scale 1/(1-p)
            hf[w][c] = v;
            sum += v; sumsq += v * v;
        }
        #pragma unroll
        for (int off = 16; off > 0; off >>= 1) {
            sum   += __shfl_xor(sum,   off, 32);
            sumsq += __shfl_xor(sumsq, off, 32);
        }
        float mean = sum * (1.f / HID);
        float var  = sumsq * (1.f / HID) - mean * mean;
        float rstd = rsqrtf(var + 1e-5f);
        for (int c = lane; c < HID; c += 32)
            hf[w][c] = (hf[w][c] - mean) * rstd * ln_w[c] + ln_b[c];
    }
    __syncthreads();

    for (int o = lane; o < 256; o += 32) {
        float acc = b1[o];
        const float* wr = w1 + (size_t)o * HID;
        for (int k = 0; k < HID; ++k) acc += hf[w][k] * wr[k];
        h1s[w][o] = acc;
    }
    __syncthreads();

    if (threadIdx.x < 32) {
        int row = lane >> 1, o = lane & 1;
        float acc = out_bias[o];
        const float* wr = w2 + (size_t)o * 256;
        for (int k = 0; k < 256; ++k) acc += h1s[row][k] * wr[k];
        out[(size_t)(b0 + row) * 2 + o] = 1.f / (1.f + __expf(-acc));
    }
}

// ---------------- launch ----------------
extern "C" void kernel_launch(void* const* d_in, const int* in_sizes, int n_in,
                              void* d_out, int out_size, void* d_ws, size_t ws_size,
                              hipStream_t stream) {
    const float* x        = (const float*)d_in[0];
    const float* w_ih     = (const float*)d_in[1];
    const float* w_hh     = (const float*)d_in[2];
    const float* b        = (const float*)d_in[3];
    const float* b_n      = (const float*)d_in[4];
    const float* ln_w     = (const float*)d_in[5];
    const float* ln_b     = (const float*)d_in[6];
    const float* w1       = (const float*)d_in[7];
    const float* b1       = (const float*)d_in[8];
    const float* w2       = (const float*)d_in[9];
    const float* out_bias = (const float*)d_in[10];
    float* out = (float*)d_out;

    char* ws = (char*)d_ws;
    unsigned short* xpk   = (unsigned short*)(ws);                 // 167,772,160 B
    unsigned short* wihpk = (unsigned short*)(ws + 167772160);     //     983,040 B
    unsigned short* whhpk = (unsigned short*)(ws + 168755200);     //   1,572,864 B
    float*          ig    = (float*)        (ws + 170328064);     // 1,610,612,736 B

    pack_x_kernel<<<327680, 256, 0, stream>>>(x, xpk);
    pack_w_kernel<<<1920, 256, 0, stream>>>(w_ih, wihpk, KT_IH, INF);
    pack_w_kernel<<<3072, 256, 0, stream>>>(w_hh, whhpk, KT_HH, HID);

    // ig GEMM: MTILES*12 wave-jobs (16x128 strip each) / 8 waves per block
    ig_gemm_kernel<<<24576, 256, 0, stream>>>(xpk, wihpk, b, ig);

    gru_recurrent_kernel<<<16, 512, 0, stream>>>(whhpk, ig, b_n, ln_w, ln_b,
                                                 w1, b1, w2, out_bias, out);
}
